// CTCDecoder_39359080300667
// MI455X (gfx1250) — compile-verified, matrix operand order checked by reading, not run on hardware
//
#include <hip/hip_runtime.h>
#include <hip/hip_bf16.h>
#include <stdint.h>
#include <math.h>

// Problem constants from the reference: x[B=4096, T=128, C=96] fp32, blank = C-1.
#define CT_T     128
#define CT_C     96
#define CT_BLANK (CT_C - 1)
#define ROW_DW   (CT_T * CT_C)                    // 12288 dwords per batch row
#define ROW_DW_PAD (ROW_DW + 2 * (ROW_DW / 32))   // TDM pads 2 dwords every 32 -> 13056

typedef __attribute__((ext_vector_type(16))) _Float16     v16h;
typedef __attribute__((ext_vector_type(8)))  float        v8f;
typedef __attribute__((ext_vector_type(4)))  unsigned int v4u;
typedef __attribute__((ext_vector_type(8)))  int          v8i;
typedef __attribute__((ext_vector_type(4)))  int          v4i;

// LDS index after TDM padding: +2 dwords after every 32 dwords stored.
// Per-timestep row stride becomes 102 dwords -> conflict-free across a wave32.
__device__ __forceinline__ int padidx(int i) { return i + ((i >> 5) << 1); }

__global__ __launch_bounds__(128)
void ctc_greedy_decode_kernel(const float* __restrict__ x,
                              float* __restrict__ out,
                              int Bcnt) {
  __shared__ float s_row[ROW_DW_PAD];   // staged batch row (padded)
  __shared__ int   s_best[CT_T];
  __shared__ float s_mlp[CT_T];
  __shared__ int   s_scan[CT_T];
  __shared__ int   s_out[CT_T];

  const int b = blockIdx.x;
  const int t = threadIdx.x;
  const float* src = x + (size_t)b * ROW_DW;

  // ---- Stage the whole batch row into LDS ------------------------------
#if __has_builtin(__builtin_amdgcn_tensor_load_to_lds)
  if (t < 32) {  // one wave issues the TDM descriptor (EXEC-independent DMA)
    unsigned long long ga = (unsigned long long)(uintptr_t)src;
    unsigned lds_off = (unsigned)(uintptr_t)(&s_row[0]);  // low 32 bits = LDS byte offset

    // D# group 0: count=1 | lds_addr | global_addr[56:0] | type=2
    v4u g0;
    g0[0] = 1u;
    g0[1] = lds_off;
    g0[2] = (unsigned)(ga & 0xFFFFFFFFull);
    g0[3] = (unsigned)((ga >> 32) & 0x01FFFFFFull) | (2u << 30);

    // D# group 1: data_size=4B, pad_enable, pad_interval=32dw, pad_amount=2dw,
    // 1 row of 12288 elems: tensor_dim0=tile_dim0=12288, tensor_dim1=tile_dim1=1.
    v8i g1;
    g1[0] = (2 << 16) | (1 << 20) | (4 << 22) | (1 << 25);
    g1[1] = (int)(0x3000u << 16);   // tensor_dim0[15:0]=12288 at bits[63:48]
    g1[2] = (int)(1u << 16);        // tensor_dim0 hi=0; tensor_dim1[15:0]=1 at [95:80]
    g1[3] = (int)(0x3000u << 16);   // tensor_dim1 hi=0; tile_dim0=12288 at [127:112]
    g1[4] = 1;                      // tile_dim1=1; tile_dim2=0
    g1[5] = ROW_DW;                 // tensor_dim0_stride = 12288
    g1[6] = (int)(0x3000u << 16);   // tensor_dim1_stride[15:0] = 12288
    g1[7] = 0;

    v4i g2 = {0, 0, 0, 0};
    v4i g3 = {0, 0, 0, 0};
    v8i g4 = {0, 0, 0, 0, 0, 0, 0, 0};  // extra group in clang-23 6-arg form
    __builtin_amdgcn_tensor_load_to_lds(g0, g1, g2, g3, g4, 0);
    __builtin_amdgcn_s_wait_tensorcnt(0);
  }
#else
  // Fallback: coalesced cooperative load into the same padded layout.
  for (int i = t; i < ROW_DW; i += CT_T)
    s_row[padidx(i)] = src[i];
#endif
  __syncthreads();

  // ---- Per-timestep argmax / max over C=96 (log is monotone) ----------
  float maxv = s_row[padidx(t * CT_C)];
  int   arg  = 0;
  #pragma unroll
  for (int k = 1; k < CT_C; ++k) {
    float v = s_row[padidx(t * CT_C + k)];
    if (v > maxv) { maxv = v; arg = k; }   // strict '>' keeps first max (jnp.argmax)
  }
  float mlp = logf(maxv + 1e-7f);

  s_best[t] = arg;
  s_mlp[t]  = mlp;
  s_out[t]  = -1;
  __syncthreads();

  // ---- CTC compaction: keep = (best != prev) & (best != blank) --------
  int prevc = (t == 0) ? -1 : s_best[t - 1];
  int keep  = (arg != prevc && arg != CT_BLANK) ? 1 : 0;
  s_scan[t] = keep;
  __syncthreads();

  // Hillis-Steele inclusive scan over 128 entries.
  for (int off = 1; off < CT_T; off <<= 1) {
    int add = (t >= off) ? s_scan[t - off] : 0;
    __syncthreads();
    s_scan[t] += add;
    __syncthreads();
  }

  if (keep) s_out[s_scan[t] - 1] = arg;   // scatter in LDS (positions < T)
  __syncthreads();

  // Prediction (int values emitted as float, exactly representable).
  out[(size_t)b * CT_T + t] = (float)s_out[t];

  // ---- Score: -sum_t max_lp, reduced on the matrix pipe ---------------
  // A holds the 128 max_lp values (4 per lane, rest 0), B = ones, fp32 acc.
  // D[M,N] = sum_K A[M,K]; total = sum over all rows = sum of 128 values.
  if (t < 32) {
    const int lane = t;
    v16h A, Bm;
    #pragma unroll
    for (int j = 0; j < 16; ++j) { A[j] = (_Float16)0.0f; Bm[j] = (_Float16)1.0f; }
    #pragma unroll
    for (int j = 0; j < 4; ++j)  A[j] = (_Float16)s_mlp[lane * 4 + j];

    v8f Cm = {};
    Cm = __builtin_amdgcn_wmma_f32_16x16x32_f16(
        /*neg_a=*/false, A, /*neg_b=*/false, Bm,
        /*c_mod=*/(short)0, Cm, /*reuse_a=*/false, /*reuse_b=*/false);

    // Lane n holds rows {0..7} (n<16) or {8..15} (n>=16) of column n%16.
    float s = Cm[0] + Cm[1] + Cm[2] + Cm[3] + Cm[4] + Cm[5] + Cm[6] + Cm[7];
    float tot = s + __shfl_xor(s, 16, 32);
    if (lane == 0) out[(size_t)Bcnt * CT_T + b] = -tot;
  }
}

extern "C" void kernel_launch(void* const* d_in, const int* in_sizes, int n_in,
                              void* d_out, int out_size, void* d_ws, size_t ws_size,
                              hipStream_t stream) {
  const float* x = (const float*)d_in[0];
  float* out = (float*)d_out;
  const int Bcnt = in_sizes[0] / (CT_T * CT_C);  // 4096
  ctc_greedy_decode_kernel<<<dim3(Bcnt), dim3(CT_T), 0, stream>>>(x, out, Bcnt);
}
